// nf_encode_3676492005531
// MI455X (gfx1250) — compile-verified, matrix operand order checked by reading, not run on hardware
//
#include <hip/hip_runtime.h>
#include <hip/hip_bf16.h>

// ---------------------------------------------------------------------------
// Problem constants (from reference)
// ---------------------------------------------------------------------------
#define TT 128
#define NN 1024
#define HH 64
#define FCD 1760      // 16 * 110
#define SS 10
#define L1 119        // T - 9
#define L2 110        // T - 18

typedef __attribute__((ext_vector_type(16))) _Float16 v16h;
typedef __attribute__((ext_vector_type(8)))  float    v8f;

// ---------------------------------------------------------------------------
// Reductions (wave32)
// ---------------------------------------------------------------------------
__device__ __forceinline__ float waveSum(float v) {
#pragma unroll
  for (int o = 16; o > 0; o >>= 1) v += __shfl_xor(v, o, 32);
  return v;
}
__device__ __forceinline__ float waveMax(float v) {
#pragma unroll
  for (int o = 16; o > 0; o >>= 1) v = fmaxf(v, __shfl_xor(v, o, 32));
  return v;
}
__device__ __forceinline__ float blockSum(float v, float* red) {
  __syncthreads();
  int lane = threadIdx.x & 31, w = threadIdx.x >> 5;
  v = waveSum(v);
  if (lane == 0) red[w] = v;
  __syncthreads();
  float r = (threadIdx.x < (blockDim.x >> 5)) ? red[threadIdx.x] : 0.0f;
  if (w == 0) r = waveSum(r);
  if (threadIdx.x == 0) red[0] = r;
  __syncthreads();
  return red[0];
}
__device__ __forceinline__ float blockMax(float v, float* red) {
  __syncthreads();
  int lane = threadIdx.x & 31, w = threadIdx.x >> 5;
  v = waveMax(v);
  if (lane == 0) red[w] = v;
  __syncthreads();
  float r = (threadIdx.x < (blockDim.x >> 5)) ? red[threadIdx.x] : -3.4e38f;
  if (w == 0) r = waveMax(r);
  if (threadIdx.x == 0) red[0] = r;
  __syncthreads();
  return red[0];
}

// ---------------------------------------------------------------------------
// K1: conv1 + relu.  node_fea is (T,N); x[n,t]=nf[t*N+n].
// y1 layout: [c1][l][n], size 8*119*1024
// ---------------------------------------------------------------------------
__global__ void conv1_kernel(const float* __restrict__ nf,
                             const float* __restrict__ w1,
                             const float* __restrict__ b1,
                             float* __restrict__ y1) {
  __shared__ float xs[TT * 32];
  __shared__ float ws[80];
  int n0 = blockIdx.x * 32;
  for (int e = threadIdx.x; e < TT * 32; e += 256) {
    int t = e >> 5, nl = e & 31;
    xs[t * 32 + nl] = nf[(size_t)t * NN + n0 + nl];
  }
  if (threadIdx.x < 80) ws[threadIdx.x] = w1[threadIdx.x];
  __syncthreads();
  for (int e = threadIdx.x; e < 8 * L1 * 32; e += 256) {
    int nl = e & 31;
    int rest = e >> 5;
    int l = rest % L1;
    int c = rest / L1;
    float s = b1[c];
#pragma unroll
    for (int k = 0; k < 10; ++k) s += xs[(l + k) * 32 + nl] * ws[c * 10 + k];
    y1[(size_t)(c * L1 + l) * NN + n0 + nl] = fmaxf(s, 0.0f);
  }
}

// ---------------------------------------------------------------------------
// BN stats over a contiguous per-channel plane -> scale/shift
// ---------------------------------------------------------------------------
__global__ void bn_stats_kernel(const float* __restrict__ x, long per_ch,
                                const float* __restrict__ g, const float* __restrict__ b,
                                float* __restrict__ scale, float* __restrict__ shift) {
  __shared__ float red[8];
  int c = blockIdx.x;
  const float* p = x + (size_t)c * per_ch;
  float s = 0.f, s2 = 0.f;
  for (long e = threadIdx.x; e < per_ch; e += blockDim.x) {
    float v = p[e]; s += v; s2 += v * v;
  }
  s = blockSum(s, red);
  s2 = blockSum(s2, red);
  if (threadIdx.x == 0) {
    float m = s / (float)per_ch;
    float var = s2 / (float)per_ch - m * m;
    float sc = g[c] * rsqrtf(var + 1e-5f);
    scale[c] = sc;
    shift[c] = b[c] - m * sc;
  }
}

// BN stats over a strided column (for z: [n][h])
__global__ void bn_stats_col_kernel(const float* __restrict__ x, int rows, int ldx,
                                    const float* __restrict__ g, const float* __restrict__ b,
                                    float* __restrict__ scale, float* __restrict__ shift) {
  __shared__ float red[8];
  int c = blockIdx.x;
  float s = 0.f, s2 = 0.f;
  for (int r = threadIdx.x; r < rows; r += blockDim.x) {
    float v = x[(size_t)r * ldx + c]; s += v; s2 += v * v;
  }
  s = blockSum(s, red);
  s2 = blockSum(s2, red);
  if (threadIdx.x == 0) {
    float m = s / (float)rows;
    float var = s2 / (float)rows - m * m;
    float sc = g[c] * rsqrtf(var + 1e-5f);
    scale[c] = sc;
    shift[c] = b[c] - m * sc;
  }
}

// ---------------------------------------------------------------------------
// Fold BN1 affine into conv2 weights:  w2s = w2*scale1[c1],
// b2eff[c2] = b2[c2] + sum_{c1,k} shift1[c1]*w2[c2,c1,k]
// ---------------------------------------------------------------------------
__global__ void fold_w2_kernel(const float* __restrict__ w2, const float* __restrict__ b2,
                               const float* __restrict__ scale1, const float* __restrict__ shift1,
                               float* __restrict__ w2s, float* __restrict__ b2eff) {
  for (int i = threadIdx.x; i < 16 * 8 * 10; i += 256) {
    int c1 = (i / 10) % 8;
    w2s[i] = w2[i] * scale1[c1];
  }
  if (threadIdx.x < 16) {
    int c2 = threadIdx.x;
    float s = b2[c2];
    for (int c1 = 0; c1 < 8; ++c1) {
      float sh = shift1[c1];
      for (int k = 0; k < 10; ++k) s += sh * w2[c2 * 80 + c1 * 10 + k];
    }
    b2eff[c2] = s;
  }
}

// ---------------------------------------------------------------------------
// K3: conv2 + relu on raw y1 (BN1 folded into weights).
// y2 layout: [k = c2*110 + l][n]
// ---------------------------------------------------------------------------
__global__ void conv2_kernel(const float* __restrict__ y1,
                             const float* __restrict__ w2s,
                             const float* __restrict__ b2eff,
                             float* __restrict__ y2) {
  __shared__ float ws[80];
  int n0 = blockIdx.x * 32;
  int c2 = blockIdx.y;
  if (threadIdx.x < 80) ws[threadIdx.x] = w2s[c2 * 80 + threadIdx.x];
  __syncthreads();
  float bb = b2eff[c2];
  for (int e = threadIdx.x; e < L2 * 32; e += 256) {
    int nl = e & 31;
    int l = e >> 5;
    float s = bb;
#pragma unroll
    for (int c1 = 0; c1 < 8; ++c1) {
      const float* base = y1 + (size_t)(c1 * L1 + l) * NN + n0 + nl;
#pragma unroll
      for (int k = 0; k < 10; ++k) s += base[(size_t)k * NN] * ws[c1 * 10 + k];
    }
    y2[(size_t)(c2 * L2 + l) * NN + n0 + nl] = fmaxf(s, 0.0f);
  }
}

// ---------------------------------------------------------------------------
// Apply BN2 + transpose-pack to fp16:  xh[n][k] = h( y2[k][n]*scale2 + shift2 )
// ---------------------------------------------------------------------------
__global__ void pack_x_kernel(const float* __restrict__ y2,
                              const float* __restrict__ scale2, const float* __restrict__ shift2,
                              _Float16* __restrict__ xh) {
  size_t e = (size_t)blockIdx.x * blockDim.x + threadIdx.x;
  if (e >= (size_t)NN * (FCD / 2)) return;
  int n = (int)(e / (FCD / 2));
  int k = 2 * (int)(e % (FCD / 2));
  int ca = k / L2, cb = (k + 1) / L2;
  float va = y2[(size_t)k * NN + n] * scale2[ca] + shift2[ca];
  float vb = y2[(size_t)(k + 1) * NN + n] * scale2[cb] + shift2[cb];
  xh[(size_t)n * FCD + k] = (_Float16)va;
  xh[(size_t)n * FCD + k + 1] = (_Float16)vb;
}

__global__ void pack_w_kernel(const float* __restrict__ w, _Float16* __restrict__ wh, int count) {
  int e = blockIdx.x * blockDim.x + threadIdx.x;
  if (e < count) wh[e] = (_Float16)w[e];
}

// ---------------------------------------------------------------------------
// WMMA fragment load: 16-bit A matrix 16x32 (MxK), row-major src, leading dim ld
// (B fragment for X @ W^T loads identically from row-major W.)
// ---------------------------------------------------------------------------
__device__ __forceinline__ v16h load_frag16(const _Float16* __restrict__ p, int ld,
                                            int row0, int k0) {
  int lane = threadIdx.x & 31;
  const _Float16* r = p + (size_t)(row0 + (lane & 15)) * ld + k0 + ((lane >> 4) << 3);
  v16h f;
#pragma unroll
  for (int v = 0; v < 8; ++v) f[v] = r[v];        // K = kg..kg+7   -> VGPR0..3
#pragma unroll
  for (int v = 0; v < 8; ++v) f[8 + v] = r[16 + v]; // K = 16+kg..   -> VGPR4..7
  return f;
}

// C = relu?( A(16x K) x B^T + bias[col] );  one wave per 16x16 tile
__global__ void wmma_gemm_kernel(const _Float16* __restrict__ A, const _Float16* __restrict__ B,
                                 const float* __restrict__ bias, float* __restrict__ C,
                                 int K, int ldc, int do_relu) {
  int tm = blockIdx.x, tn = blockIdx.y;
  v8f acc = {};
  for (int k0 = 0; k0 < K; k0 += 32) {
    v16h a = load_frag16(A, K, tm * 16, k0);
    v16h b = load_frag16(B, K, tn * 16, k0);
    acc = __builtin_amdgcn_wmma_f32_16x16x32_f16(false, a, false, b, (short)0, acc,
                                                 false, false);
  }
  int lane = threadIdx.x & 31;
  int col = tn * 16 + (lane & 15);
  float bc = bias[col];
  int rbase = tm * 16 + ((lane >> 4) << 3);
#pragma unroll
  for (int v = 0; v < 8; ++v) {
    float x = acc[v] + bc;
    if (do_relu) x = fmaxf(x, 0.0f);
    C[(size_t)(rbase + v) * ldc + col] = x;
  }
}

// ---------------------------------------------------------------------------
// Fold BN3 into fc2 halves; pack z to fp16.
// Wah[p][h] = fc2_w[p, h]*scale3[h];  Wch[p][h] = fc2_w[p, 64+h]*scale3[h]
// bias_a[p] = sum_h fc2_w[p,h]*shift3[h];  bias_c likewise.
// ---------------------------------------------------------------------------
__global__ void prep_ac_kernel(const float* __restrict__ fc2_w,
                               const float* __restrict__ scale3, const float* __restrict__ shift3,
                               const float* __restrict__ z, _Float16* __restrict__ zh,
                               _Float16* __restrict__ Wah, _Float16* __restrict__ Wch,
                               float* __restrict__ bias_a, float* __restrict__ bias_c) {
  int e = blockIdx.x * blockDim.x + threadIdx.x;
  if (e < NN * HH) zh[e] = (_Float16)z[e];
  if (e < HH * HH) {
    int p = e >> 6, h = e & 63;
    Wah[e] = (_Float16)(fc2_w[p * 128 + h] * scale3[h]);
    Wch[e] = (_Float16)(fc2_w[p * 128 + 64 + h] * scale3[h]);
  }
  if (e < HH) {
    float sa = 0.f, sc = 0.f;
    for (int h = 0; h < HH; ++h) {
      sa += fc2_w[e * 128 + h] * shift3[h];
      sc += fc2_w[e * 128 + 64 + h] * shift3[h];
    }
    bias_a[e] = sa;
    bias_c[e] = sc;
  }
}

// ---------------------------------------------------------------------------
// Pairwise logits + Gumbel-softmax over S samples + row normalization.
// One block (256 threads) per row i.  Per-thread j = tid + 256*q, q=0..3.
// ---------------------------------------------------------------------------
__global__ void pairwise_kernel(const float* __restrict__ amat, const float* __restrict__ cmat,
                                const float* __restrict__ fc2_b, const float* __restrict__ fc1_w,
                                const float* __restrict__ fc1_b, const float* __restrict__ gum,
                                float* __restrict__ out) {
  __shared__ float av[HH];
  __shared__ float w1s[HH];
  __shared__ float red[8];
  int i = blockIdx.x;
  int tid = threadIdx.x;
  if (tid < HH) {
    av[tid] = amat[(size_t)i * HH + tid] + fc2_b[tid];
    w1s[tid] = fc1_w[tid];
  }
  __syncthreads();
  float fb = fc1_b[0];

  float lg[4];
#pragma unroll
  for (int q = 0; q < 4; ++q) {
    int j = tid + q * 256;
    const float* cr = cmat + (size_t)j * HH;
    float acc = fb;
#pragma unroll
    for (int h = 0; h < HH; ++h) acc += w1s[h] * fmaxf(av[h] + cr[h], 0.0f);
    lg[q] = acc;
  }

  float psum[4] = {0.f, 0.f, 0.f, 0.f};
  for (int s = 0; s < SS; ++s) {
    const float* gr = gum + ((size_t)s * NN + i) * NN;
    if (s + 1 < SS)
      __builtin_prefetch(gum + ((size_t)(s + 1) * NN + i) * NN + tid, 0, 1);
    float tl[4];
    float lmax = -3.4e38f;
#pragma unroll
    for (int q = 0; q < 4; ++q) {
      int j = tid + q * 256;
      float u = gr[j];
      float g = -logf(-logf(u + 1e-10f) + 1e-10f);
      float t = (lg[q] + g) * 10.0f;   // /TEMP, TEMP=0.1
      tl[q] = t;
      lmax = fmaxf(lmax, t);
    }
    float bmax = blockMax(lmax, red);
    float ls = 0.f;
    float ex[4];
#pragma unroll
    for (int q = 0; q < 4; ++q) {
      ex[q] = expf(tl[q] - bmax);
      ls += ex[q];
    }
    float bs = blockSum(ls, red);
    float inv = 1.0f / bs;
#pragma unroll
    for (int q = 0; q < 4; ++q) psum[q] += ex[q] * inv;
  }

  // mean over S then row-normalize: S cancels -> psum / sum(psum)
  float ls = psum[0] + psum[1] + psum[2] + psum[3];
  float tot = blockSum(ls, red);
  float inv = 1.0f / tot;
#pragma unroll
  for (int q = 0; q < 4; ++q) {
    int j = tid + q * 256;
    out[(size_t)i * NN + j] = psum[q] * inv;
  }
}

// ---------------------------------------------------------------------------
// Launch
// ---------------------------------------------------------------------------
extern "C" void kernel_launch(void* const* d_in, const int* in_sizes, int n_in,
                              void* d_out, int out_size, void* d_ws, size_t ws_size,
                              hipStream_t stream) {
  const float* node_fea = (const float*)d_in[0];
  const float* gumbel_u = (const float*)d_in[1];
  const float* conv1_w  = (const float*)d_in[2];
  const float* conv1_b  = (const float*)d_in[3];
  const float* conv2_w  = (const float*)d_in[4];
  const float* conv2_b  = (const float*)d_in[5];
  const float* bn1_g    = (const float*)d_in[6];
  const float* bn1_b    = (const float*)d_in[7];
  const float* bn2_g    = (const float*)d_in[8];
  const float* bn2_b    = (const float*)d_in[9];
  const float* bn3_g    = (const float*)d_in[10];
  const float* bn3_b    = (const float*)d_in[11];
  const float* fc_w     = (const float*)d_in[12];
  const float* fc_b     = (const float*)d_in[13];
  const float* fc2_w    = (const float*)d_in[14];
  const float* fc2_b    = (const float*)d_in[15];
  const float* fc1_w    = (const float*)d_in[16];
  const float* fc1_b    = (const float*)d_in[17];
  float* out = (float*)d_out;

  char* base = (char*)d_ws;
  size_t off = 0;
  auto alloc = [&](size_t bytes) -> char* {
    char* p = base + off;
    off += (bytes + 255) & ~(size_t)255;
    return p;
  };

  float*    y1     = (float*)   alloc((size_t)8 * L1 * NN * 4);
  float*    y2     = (float*)   alloc((size_t)FCD * NN * 4);
  _Float16* xh     = (_Float16*)alloc((size_t)NN * FCD * 2);
  _Float16* fcwh   = (_Float16*)alloc((size_t)HH * FCD * 2);
  float*    z      = (float*)   alloc((size_t)NN * HH * 4);
  _Float16* zh     = (_Float16*)alloc((size_t)NN * HH * 2);
  _Float16* Wah    = (_Float16*)alloc((size_t)HH * HH * 2);
  _Float16* Wch    = (_Float16*)alloc((size_t)HH * HH * 2);
  float*    bias_a = (float*)   alloc(HH * 4);
  float*    bias_c = (float*)   alloc(HH * 4);
  float*    amat   = (float*)   alloc((size_t)NN * HH * 4);
  float*    cmat   = (float*)   alloc((size_t)NN * HH * 4);
  float*    scale1 = (float*)   alloc(8 * 4);
  float*    shift1 = (float*)   alloc(8 * 4);
  float*    scale2 = (float*)   alloc(16 * 4);
  float*    shift2 = (float*)   alloc(16 * 4);
  float*    scale3 = (float*)   alloc(HH * 4);
  float*    shift3 = (float*)   alloc(HH * 4);
  float*    w2s    = (float*)   alloc(1280 * 4);
  float*    b2eff  = (float*)   alloc(16 * 4);

  // 1. conv1 + relu
  conv1_kernel<<<NN / 32, 256, 0, stream>>>(node_fea, conv1_w, conv1_b, y1);
  // 2. BN1 stats
  bn_stats_kernel<<<8, 256, 0, stream>>>(y1, (long)L1 * NN, bn1_g, bn1_b, scale1, shift1);
  // 3. fold BN1 into conv2 weights
  fold_w2_kernel<<<1, 256, 0, stream>>>(conv2_w, conv2_b, scale1, shift1, w2s, b2eff);
  // 4. conv2 + relu
  conv2_kernel<<<dim3(NN / 32, 16), 256, 0, stream>>>(y1, w2s, b2eff, y2);
  // 5. BN2 stats
  bn_stats_kernel<<<16, 256, 0, stream>>>(y2, (long)L2 * NN, bn2_g, bn2_b, scale2, shift2);
  // 6. apply BN2 + transpose pack to fp16
  {
    size_t total = (size_t)NN * (FCD / 2);
    pack_x_kernel<<<(unsigned)((total + 255) / 256), 256, 0, stream>>>(y2, scale2, shift2, xh);
  }
  // 7. pack fc_w to fp16
  pack_w_kernel<<<(HH * FCD + 255) / 256, 256, 0, stream>>>(fc_w, fcwh, HH * FCD);
  // 8. fc GEMM (1024 x 1760 x 64) with WMMA, + bias + relu
  wmma_gemm_kernel<<<dim3(NN / 16, HH / 16), 32, 0, stream>>>(xh, fcwh, fc_b, z, FCD, HH, 1);
  // 9. BN3 stats (per column of z)
  bn_stats_col_kernel<<<HH, 256, 0, stream>>>(z, NN, HH, bn3_g, bn3_b, scale3, shift3);
  // 10. fold BN3 into fc2 halves, pack z
  prep_ac_kernel<<<(NN * HH + 255) / 256, 256, 0, stream>>>(fc2_w, scale3, shift3, z, zh,
                                                            Wah, Wch, bias_a, bias_c);
  // 11/12. a = zh @ Wah^T + bias_a ; c = zh @ Wch^T + bias_c  (WMMA)
  wmma_gemm_kernel<<<dim3(NN / 16, HH / 16), 32, 0, stream>>>(zh, Wah, bias_a, amat, HH, HH, 0);
  wmma_gemm_kernel<<<dim3(NN / 16, HH / 16), 32, 0, stream>>>(zh, Wch, bias_c, cmat, HH, HH, 0);
  // 13. pairwise + gumbel softmax + row normalize
  pairwise_kernel<<<NN, 256, 0, stream>>>(amat, cmat, fc2_b, fc1_w, fc1_b, gumbel_u, out);
}
